// DetailTransformer_24266565222641
// MI455X (gfx1250) — matrix operand order repaired and checked by reading, not compiled, not run on hardware
//
#include <hip/hip_runtime.h>

// ---------------------------------------------------------------------------
// CDNA5 (gfx1250) DetailTransformer forward.
// All GEMMs run on v_wmma_f32_16x16x32_bf16 (fp32->bf16 on the fly). LDS tiles
// are stored directly in WMMA fragment layout (two ds_load_b128 per operand
// per lane), double-buffered so next-tile global loads overlap the WMMAs.
// FULL-template instantiations carry zero bounds logic (no spill pressure).
// wave32; 128-thread blocks = 4 waves; 64x64 C tiles, K-step 32.
// ---------------------------------------------------------------------------

typedef __attribute__((ext_vector_type(16))) __bf16 v16bf;
typedef __attribute__((ext_vector_type(8)))  __bf16 v8bf;
typedef __attribute__((ext_vector_type(8)))  float  v8f;

#define WMMA_BF16(A_, B_, C_) \
  __builtin_amdgcn_wmma_f32_16x16x32_bf16(false, (A_), false, (B_), (short)0, (C_), false, false)

constexpr int BM = 64, BN = 64, BK = 32;
constexpr int T_SEQ = 512, QS = 1536, DH = 64;
constexpr int GS = 1024, NPOS = 1023;          // rel-pos table
constexpr int CH = 8;                           // (b,h) pairs per attention chunk
constexpr int XIN_LD = 160;                     // padded encoder input width (even)
// Fragment region: 32 lanes x 8 dwords, +16B stagger per 8-lane group (bank spread)
constexpr int FRAG_DW = 272;                    // dwords per fragment (1088 B)
constexpr int FRAGS   = 4 * FRAG_DW;            // one tile set (4 fragments)

__device__ __forceinline__ float prelu_f(float x, float a) { return x >= 0.f ? x : a * x; }
__device__ __forceinline__ v8f vzero8() { v8f z = {0.f,0.f,0.f,0.f,0.f,0.f,0.f,0.f}; return z; }

__device__ __forceinline__ unsigned pack2(float a, float b) {
#if __has_builtin(__builtin_amdgcn_cvt_pk_bf16_f32)
  auto r = __builtin_amdgcn_cvt_pk_bf16_f32(a, b);
  return __builtin_bit_cast(unsigned, r);
#else
  union { __bf16 h[2]; unsigned u; } pk;
  pk.h[0] = (__bf16)a; pk.h[1] = (__bf16)b;
  return pk.u;
#endif
}

__device__ __forceinline__ int lane_off_dw(int lane) {   // staggered per-lane base
  return lane * 8 + ((lane >> 3) << 2);
}

// Load one 16x16x32 operand fragment: two ds_load_b128 per lane.
__device__ __forceinline__ v16bf frag_ld(const unsigned* s, int frag, int lane) {
  const v8bf* p = (const v8bf*)(s + frag * FRAG_DW + lane_off_dw(lane));
  v8bf lo = p[0], hi = p[1];
  return __builtin_shufflevector(lo, hi, 0,1,2,3,4,5,6,7,8,9,10,11,12,13,14,15);
}

// Fragment-layout scatter positions (ISA 7.12.2).
__device__ __forceinline__ void fstore_a(unsigned* s, int r, int k, unsigned d) {
  int lt = (r & 15) + 16 * ((k >> 3) & 1);
  int e  = (k & 7) | ((k >> 4) << 3);
  s[(r >> 4) * FRAG_DW + lane_off_dw(lt) + (e >> 1)] = d;
}
__device__ __forceinline__ void fstore_b(unsigned* s, int n, int k, unsigned d) {
  int lt = (n & 15) + 16 * ((k >> 3) & 1);
  int e  = (k & 7) | ((k >> 4) << 3);
  s[(n >> 4) * FRAG_DW + lane_off_dw(lt) + (e >> 1)] = d;
}

// ---- staging: bf16x2 dwords directly into fragment layout ------------------
// FULL=true: statically in-bounds, lda even, no guards at all.
template <bool FULL>
__device__ __forceinline__ void stage_a(unsigned* s, const float* __restrict__ A, size_t lda,
                                        int m0, int k0, int Mlim, int Klim, int tid) {
  if (FULL) {
#pragma unroll
    for (int it = 0; it < 8; ++it) {
      int idx = tid + it * 128;
      int r = idx >> 4, k = (idx & 15) * 2;
      float2 v = *(const float2*)(A + (size_t)(m0 + r) * lda + k0 + k);
      fstore_a(s, r, k, pack2(v.x, v.y));
    }
  } else {
#pragma unroll
    for (int it = 0; it < 8; ++it) {
      int idx = tid + it * 128;
      int r = idx >> 4, k = (idx & 15) * 2;
      int gm = m0 + r;
      float v0 = 0.f, v1 = 0.f;
      if (gm < Mlim) {
        const float* p = A + (size_t)gm * lda + k0 + k;
        if (k0 + k     < Klim) v0 = p[0];
        if (k0 + k + 1 < Klim) v1 = p[1];
      }
      fstore_a(s, r, k, pack2(v0, v1));
    }
  }
}

// B from row-major src[k][n] (pairs stride ldw apart).
template <bool FULL>
__device__ __forceinline__ void stage_b(unsigned* s, const float* __restrict__ W, size_t ldw,
                                        int n0, int k0, int Klim, int Nlim, int tid) {
  if (FULL) {
#pragma unroll
    for (int it = 0; it < 8; ++it) {
      int idx = tid + it * 128;
      int k = (idx >> 6) * 2, n = idx & 63;
      const float* p = W + (size_t)(k0 + k) * ldw + n0 + n;
      fstore_b(s, n, k, pack2(p[0], p[ldw]));
    }
  } else {
#pragma unroll
    for (int it = 0; it < 8; ++it) {
      int idx = tid + it * 128;
      int k = (idx >> 6) * 2, n = idx & 63;
      int gn = n0 + n;
      float v0 = 0.f, v1 = 0.f;
      if (gn < Nlim) {
        if (k0 + k     < Klim) v0 = W[(size_t)(k0 + k) * ldw + gn];
        if (k0 + k + 1 < Klim) v1 = W[(size_t)(k0 + k + 1) * ldw + gn];
      }
      fstore_b(s, n, k, pack2(v0, v1));
    }
  }
}

// B from transposed source src[n][k] (pair contiguous: float2).
template <bool FULL>
__device__ __forceinline__ void stage_bt(unsigned* s, const float* __restrict__ Wt, size_t ldw,
                                         int n0, int k0, int Nlim, int Klim, int tid) {
  if (FULL) {
#pragma unroll
    for (int it = 0; it < 8; ++it) {
      int idx = tid + it * 128;
      int n = idx >> 4, k = (idx & 15) * 2;
      float2 v = *(const float2*)(Wt + (size_t)(n0 + n) * ldw + k0 + k);
      fstore_b(s, n, k, pack2(v.x, v.y));
    }
  } else {
#pragma unroll
    for (int it = 0; it < 8; ++it) {
      int idx = tid + it * 128;
      int n = idx >> 4, k = (idx & 15) * 2;
      int gn = n0 + n;
      float v0 = 0.f, v1 = 0.f;
      if (gn < Nlim) {
        const float* p = Wt + (size_t)gn * ldw + k0 + k;
        if (k0 + k     < Klim) v0 = p[0];
        if (k0 + k + 1 < Klim) v1 = p[1];
      }
      fstore_b(s, n, k, pack2(v0, v1));
    }
  }
}

// ---------------------------------------------------------------------------
// Generic GEMM: C = act(A[M,K] @ W[K,N] + bias) (+ resid). act: 0 none,
// 1 relu, 2 prelu(*alpha_ptr). lda/Ka: A row stride & padded A width.
// ---------------------------------------------------------------------------
template <bool FULL>
__global__ __launch_bounds__(128, 1)
void k_gemm_t(const float* __restrict__ A, const float* __restrict__ W,
              const float* __restrict__ bias, const float* __restrict__ resid,
              const float* __restrict__ alpha_ptr, float* __restrict__ C,
              int M, int N, int K, int lda, int Ka, int act) {
  __shared__ unsigned sAu[2 * FRAGS];
  __shared__ unsigned sBu[2 * FRAGS];
  int m0 = blockIdx.y * BM, n0 = blockIdx.x * BN;
  int tid = threadIdx.x, lane = tid & 31, wave = tid >> 5;
  v8f acc[4];
#pragma unroll
  for (int t = 0; t < 4; ++t) acc[t] = vzero8();

  int nk = (K + BK - 1) / BK;
  stage_a<FULL>(sAu, A, lda, m0, 0, M, Ka, tid);
  stage_b<FULL>(sBu, W, N, n0, 0, K, N, tid);
  int cur = 0;
  for (int t = 0; t < nk; ++t) {
    __syncthreads();
    const unsigned* sa = sAu + cur * FRAGS;
    const unsigned* sb = sBu + cur * FRAGS;
    v16bf af = frag_ld(sa, wave, lane);
#pragma unroll
    for (int nt = 0; nt < 4; ++nt) {
      v16bf bf = frag_ld(sb, nt, lane);
      acc[nt] = WMMA_BF16(af, bf, acc[nt]);
    }
    if (t + 1 < nk) {
      int k0 = (t + 1) * BK;
      stage_a<FULL>(sAu + (cur ^ 1) * FRAGS, A, lda, m0, k0, M, Ka, tid);
      stage_b<FULL>(sBu + (cur ^ 1) * FRAGS, W, N, n0, k0, K, N, tid);
    }
    cur ^= 1;
  }

  float alpha = (act == 2 && alpha_ptr) ? alpha_ptr[0] : 0.f;
  bool fullC = FULL || ((m0 + BM <= M) && (n0 + BN <= N));
  int half = lane >> 4, col = lane & 15;
#pragma unroll
  for (int nt = 0; nt < 4; ++nt) {
#pragma unroll
    for (int v = 0; v < 8; ++v) {
      int m = m0 + wave * 16 + v + 8 * half;
      int n = n0 + nt * 16 + col;
      if (FULL || fullC || (m < M && n < N)) {
        float val = acc[nt][v];
        if (bias)  val += bias[n];
        if (resid) val += resid[(size_t)m * N + n];
        if (act == 1)      val = val > 0.f ? val : 0.f;
        else if (act == 2) val = prelu_f(val, alpha);
        C[(size_t)m * N + n] = val;
      }
    }
  }
}

// ---------------------------------------------------------------------------
// MoE blended GEMM: out = act( sum_e gw[m,e] * (Z @ W[e] + b[e]) )
// Flattened (expert x k-tile) software pipeline.
// ---------------------------------------------------------------------------
template <bool FULL>
__global__ __launch_bounds__(128, 1)
void k_moe_t(const float* __restrict__ Z, const float* __restrict__ W,
             const float* __restrict__ Bml, const float* __restrict__ gw,
             const float* __restrict__ alpha_ptr, int act,
             float* __restrict__ C, int M, int N, int K, int E) {
  __shared__ unsigned sAu[2 * FRAGS];
  __shared__ unsigned sBu[2 * FRAGS];
  __shared__ float    sGW[BM * 8];
  int m0 = blockIdx.y * BM, n0 = blockIdx.x * BN;
  int tid = threadIdx.x, lane = tid & 31, wave = tid >> 5;
  int half = lane >> 4, col = lane & 15;

  for (int idx = tid; idx < BM * E; idx += 128) {
    int r = idx / E, e = idx % E;
    int gm = m0 + r;
    sGW[idx] = (FULL || gm < M) ? gw[(size_t)gm * E + e] : 0.f;
  }

  v8f fin[4], acc[4];
#pragma unroll
  for (int t = 0; t < 4; ++t) { fin[t] = vzero8(); acc[t] = vzero8(); }

  int nk = (K + BK - 1) / BK;
  int steps = E * nk;
  stage_a<FULL>(sAu, Z, K, m0, 0, M, K, tid);
  stage_b<FULL>(sBu, W, N, n0, 0, K, N, tid);
  int cur = 0;
  for (int s = 0; s < steps; ++s) {
    int e = s / nk, t = s % nk;
    __syncthreads();
    const unsigned* sa = sAu + cur * FRAGS;
    const unsigned* sb = sBu + cur * FRAGS;
    v16bf af = frag_ld(sa, wave, lane);
#pragma unroll
    for (int nt = 0; nt < 4; ++nt) {
      v16bf bf = frag_ld(sb, nt, lane);
      acc[nt] = WMMA_BF16(af, bf, acc[nt]);
    }
    if (s + 1 < steps) {
      int en = (s + 1) / nk, tn = (s + 1) % nk;
      int k0 = tn * BK;
      stage_a<FULL>(sAu + (cur ^ 1) * FRAGS, Z, K, m0, k0, M, K, tid);
      stage_b<FULL>(sBu + (cur ^ 1) * FRAGS, W + (size_t)en * K * N, N, n0, k0, K, N, tid);
    }
    if (t == nk - 1) {        // fold expert e into the gated accumulator
#pragma unroll
      for (int nt = 0; nt < 4; ++nt) {
#pragma unroll
        for (int v = 0; v < 8; ++v) {
          int ml = wave * 16 + v + 8 * half;
          int n  = n0 + nt * 16 + col;
          float b = (FULL || n < N) ? Bml[(size_t)e * N + n] : 0.f;
          fin[nt][v] += (acc[nt][v] + b) * sGW[ml * E + e];
        }
        acc[nt] = vzero8();
      }
    }
    cur ^= 1;
  }

  float alpha = (act == 2 && alpha_ptr) ? alpha_ptr[0] : 0.f;
  bool fullC = FULL || ((m0 + BM <= M) && (n0 + BN <= N));
#pragma unroll
  for (int nt = 0; nt < 4; ++nt) {
#pragma unroll
    for (int v = 0; v < 8; ++v) {
      int m = m0 + wave * 16 + v + 8 * half;
      int n = n0 + nt * 16 + col;
      if (FULL || fullC || (m < M && n < N)) {
        float val = fin[nt][v];
        if (act == 2) val = prelu_f(val, alpha);
        C[(size_t)m * N + n] = val;
      }
    }
  }
}

// ---------------------------------------------------------------------------
// G = Q @ R^T per (b,h):  G[zloc,i,p] = sum_d Q[i,d] * r[p,d]
// ---------------------------------------------------------------------------
__global__ __launch_bounds__(128, 1)
void k_bd_gemm(const float* __restrict__ qkv, const float* __restrict__ r,
               float* __restrict__ G, int bh_base) {
  __shared__ unsigned sAu[2 * FRAGS];
  __shared__ unsigned sBu[2 * FRAGS];
  int zloc = blockIdx.z, bh = bh_base + zloc;
  int b = bh >> 3, h = bh & 7;
  int m0 = blockIdx.y * BM, n0 = blockIdx.x * BN;
  int tid = threadIdx.x, lane = tid & 31, wave = tid >> 5;
  const float* Qb = qkv + (size_t)b * T_SEQ * QS + h * DH;
  v8f acc[4];
#pragma unroll
  for (int t = 0; t < 4; ++t) acc[t] = vzero8();

  stage_a<true>(sAu, Qb, QS, m0, 0, T_SEQ, DH, tid);
  stage_bt<false>(sBu, r, DH, n0, 0, NPOS, DH, tid);
  int cur = 0;
  for (int t = 0; t < 2; ++t) {
    __syncthreads();
    const unsigned* sa = sAu + cur * FRAGS;
    const unsigned* sb = sBu + cur * FRAGS;
    v16bf af = frag_ld(sa, wave, lane);
#pragma unroll
    for (int nt = 0; nt < 4; ++nt) {
      v16bf bf = frag_ld(sb, nt, lane);
      acc[nt] = WMMA_BF16(af, bf, acc[nt]);
    }
    if (t == 0) {
      stage_a<true>(sAu + FRAGS, Qb, QS, m0, BK, T_SEQ, DH, tid);
      stage_bt<false>(sBu + FRAGS, r, DH, n0, BK, NPOS, DH, tid);
    }
    cur ^= 1;
  }
  int half = lane >> 4, col = lane & 15;
#pragma unroll
  for (int nt = 0; nt < 4; ++nt) {
#pragma unroll
    for (int v = 0; v < 8; ++v) {
      int m = m0 + wave * 16 + v + 8 * half;
      int n = n0 + nt * 16 + col;
      if (n < NPOS)
        G[((size_t)zloc * T_SEQ + m) * GS + n] = acc[nt][v];
    }
  }
}

// ---------------------------------------------------------------------------
// scores[zloc,i,j] = Q_i . K_j + G[zloc, i, j-i+511]
// ---------------------------------------------------------------------------
__global__ __launch_bounds__(128, 1)
void k_attn_scores(const float* __restrict__ qkv, const float* __restrict__ G,
                   float* __restrict__ S, int bh_base) {
  __shared__ unsigned sAu[2 * FRAGS];
  __shared__ unsigned sBu[2 * FRAGS];
  int zloc = blockIdx.z, bh = bh_base + zloc;
  int b = bh >> 3, h = bh & 7;
  int m0 = blockIdx.y * BM, n0 = blockIdx.x * BN;
  int tid = threadIdx.x, lane = tid & 31, wave = tid >> 5;
  const float* Qb = qkv + (size_t)b * T_SEQ * QS + h * DH;
  const float* Kb = qkv + (size_t)b * T_SEQ * QS + 512 + h * DH;
  v8f acc[4];
#pragma unroll
  for (int t = 0; t < 4; ++t) acc[t] = vzero8();

  stage_a<true>(sAu, Qb, QS, m0, 0, T_SEQ, DH, tid);
  stage_bt<true>(sBu, Kb, QS, n0, 0, T_SEQ, DH, tid);
  int cur = 0;
  for (int t = 0; t < 2; ++t) {
    __syncthreads();
    const unsigned* sa = sAu + cur * FRAGS;
    const unsigned* sb = sBu + cur * FRAGS;
    v16bf af = frag_ld(sa, wave, lane);
#pragma unroll
    for (int nt = 0; nt < 4; ++nt) {
      v16bf bf = frag_ld(sb, nt, lane);
      acc[nt] = WMMA_BF16(af, bf, acc[nt]);
    }
    if (t == 0) {
      stage_a<true>(sAu + FRAGS, Qb, QS, m0, BK, T_SEQ, DH, tid);
      stage_bt<true>(sBu + FRAGS, Kb, QS, n0, BK, T_SEQ, DH, tid);
    }
    cur ^= 1;
  }
  int half = lane >> 4, col = lane & 15;
#pragma unroll
  for (int nt = 0; nt < 4; ++nt) {
#pragma unroll
    for (int v = 0; v < 8; ++v) {
      int m = m0 + wave * 16 + v + 8 * half;
      int n = n0 + nt * 16 + col;
      int rel = n - m + (T_SEQ - 1);      // always in [0, 1022]
      float val = acc[nt][v] + G[((size_t)zloc * T_SEQ + m) * GS + rel];
      S[((size_t)zloc * T_SEQ + m) * T_SEQ + n] = val;
    }
  }
}

// ---------------------------------------------------------------------------
// O[b,i, h*64+n] = sum_j softmax(S)[zloc,i,j] * V[j,n]
// ---------------------------------------------------------------------------
__global__ __launch_bounds__(128, 1)
void k_attn_av(const float* __restrict__ S, const float* __restrict__ qkv,
               float* __restrict__ O, int bh_base) {
  __shared__ unsigned sAu[2 * FRAGS];
  __shared__ unsigned sBu[2 * FRAGS];
  int zloc = blockIdx.z, bh = bh_base + zloc;
  int b = bh >> 3, h = bh & 7;
  int m0 = blockIdx.y * BM;
  int tid = threadIdx.x, lane = tid & 31, wave = tid >> 5;
  const float* Vb = qkv + (size_t)b * T_SEQ * QS + 1024 + h * DH;
  const float* Sb = S + (size_t)zloc * T_SEQ * T_SEQ;
  v8f acc[4];
#pragma unroll
  for (int t = 0; t < 4; ++t) acc[t] = vzero8();

  stage_a<true>(sAu, Sb, T_SEQ, m0, 0, T_SEQ, T_SEQ, tid);
  stage_b<true>(sBu, Vb, QS, 0, 0, T_SEQ, DH, tid);
  int cur = 0;
  int nk = T_SEQ / BK;
  for (int t = 0; t < nk; ++t) {
    __syncthreads();
    const unsigned* sa = sAu + cur * FRAGS;
    const unsigned* sb = sBu + cur * FRAGS;
    v16bf af = frag_ld(sa, wave, lane);
#pragma unroll
    for (int nt = 0; nt < 4; ++nt) {
      v16bf bf = frag_ld(sb, nt, lane);
      acc[nt] = WMMA_BF16(af, bf, acc[nt]);
    }
    if (t + 1 < nk) {
      int k0 = (t + 1) * BK;
      stage_a<true>(sAu + (cur ^ 1) * FRAGS, Sb, T_SEQ, m0, k0, T_SEQ, T_SEQ, tid);
      stage_b<true>(sBu + (cur ^ 1) * FRAGS, Vb, QS, 0, k0, T_SEQ, DH, tid);
    }
    cur ^= 1;
  }
  int half = lane >> 4, col = lane & 15;
#pragma unroll
  for (int nt = 0; nt < 4; ++nt) {
#pragma unroll
    for (int v = 0; v < 8; ++v) {
      int m = m0 + wave * 16 + v + 8 * half;
      int n = nt * 16 + col;                   // < 64
      O[((size_t)b * T_SEQ + m) * 512 + h * DH + n] = acc[nt][v];
    }
  }
}

// ---------------------------------------------------------------------------
// Small VALU kernels
// ---------------------------------------------------------------------------
__global__ void k_build_xin(const float* __restrict__ motion, const float* __restrict__ phase,
                            const int* __restrict__ mt, float* __restrict__ xin) {
  int row = blockIdx.x;            // b*T + t
  int t = row & (T_SEQ - 1);
  int c = threadIdx.x;
  if (c < 128) {
    xin[(size_t)row * XIN_LD + c] = motion[(size_t)row * 128 + c];
  } else if (c == 128) {
    float m = (t < 10 || t == T_SEQ - 1) ? 1.f : 0.f;
    if (t == mt[0] || t == mt[1] || t == mt[2] || t == mt[3]) m = 1.f;
    xin[(size_t)row * XIN_LD + 128] = m;
  } else if (c < 145) {
    xin[(size_t)row * XIN_LD + c] = phase[(size_t)row * 16 + (c - 129)];
  } else if (c < XIN_LD) {
    xin[(size_t)row * XIN_LD + c] = 0.f;       // zero pad -> fast-path staging
  }
}

__global__ void k_relpos(const float* __restrict__ w1, const float* __restrict__ b1,
                         const float* __restrict__ a_ptr, const float* __restrict__ w2,
                         const float* __restrict__ b2, float* __restrict__ r) {
  int p = blockIdx.x;              // 0..1022
  int d = threadIdx.x;             // 0..63
  float pos = (float)(p - (T_SEQ - 1));
  float a = a_ptr[0];
  __shared__ float hsh[64];
  hsh[d] = prelu_f(pos * w1[d] + b1[d], a);
  __syncthreads();
  float acc = b2[d];
#pragma unroll 8
  for (int hh = 0; hh < 64; ++hh) acc += hsh[hh] * w2[hh * 64 + d];
  r[(size_t)p * 64 + d] = acc;
}

__global__ void k_layernorm(const float* __restrict__ X, const float* __restrict__ g,
                            const float* __restrict__ b, float* __restrict__ Y, int N) {
  int row = blockIdx.x;
  const float* x = X + (size_t)row * N;
  float* y = Y + (size_t)row * N;
  __shared__ float red[8];
  float s = 0.f, s2 = 0.f;
  for (int i = threadIdx.x; i < N; i += blockDim.x) { float v = x[i]; s += v; s2 += v * v; }
  for (int off = 16; off > 0; off >>= 1) { s += __shfl_down(s, off, 32); s2 += __shfl_down(s2, off, 32); }
  int lane = threadIdx.x & 31, wave = threadIdx.x >> 5;
  if (lane == 0) { red[wave] = s; red[4 + wave] = s2; }
  __syncthreads();
  if (threadIdx.x == 0) {
    float ts = 0.f, ts2 = 0.f;
    for (int w = 0; w < 4; ++w) { ts += red[w]; ts2 += red[4 + w]; }
    red[0] = ts; red[4] = ts2;
  }
  __syncthreads();
  float mean = red[0] / N;
  float var  = red[4] / N - mean * mean;
  float inv  = rsqrtf(var + 1e-5f);
  for (int i = threadIdx.x; i < N; i += blockDim.x)
    y[i] = (x[i] - mean) * inv * g[i] + b[i];
}

__global__ void k_softmax_rows(float* __restrict__ S, int cols, float scale) {
  int row = blockIdx.x;
  float* x = S + (size_t)row * cols;
  __shared__ float rmax[4], rsum[4];
  float mx = -3.4e38f;
  for (int i = threadIdx.x; i < cols; i += blockDim.x) mx = fmaxf(mx, x[i] * scale);
  for (int off = 16; off > 0; off >>= 1) mx = fmaxf(mx, __shfl_xor(mx, off, 32));
  int lane = threadIdx.x & 31, wave = threadIdx.x >> 5;
  if (lane == 0) rmax[wave] = mx;
  __syncthreads();
  mx = fmaxf(fmaxf(rmax[0], rmax[1]), fmaxf(rmax[2], rmax[3]));
  float s = 0.f;
  for (int i = threadIdx.x; i < cols; i += blockDim.x) {
    float e = __expf(x[i] * scale - mx);
    x[i] = e; s += e;
  }
  for (int off = 16; off > 0; off >>= 1) s += __shfl_xor(s, off, 32);
  if (lane == 0) rsum[wave] = s;
  __syncthreads();
  float inv = 1.f / (rsum[0] + rsum[1] + rsum[2] + rsum[3]);
  for (int i = threadIdx.x; i < cols; i += blockDim.x) x[i] *= inv;
}

__global__ void k_softmax8(float* __restrict__ X, int M) {
  int t = blockIdx.x * blockDim.x + threadIdx.x;
  if (t >= M) return;
  float* x = X + (size_t)t * 8;
  float mx = x[0];
#pragma unroll
  for (int i = 1; i < 8; ++i) mx = fmaxf(mx, x[i]);
  float s = 0.f;
#pragma unroll
  for (int i = 0; i < 8; ++i) { float e = __expf(x[i] - mx); x[i] = e; s += e; }
  float inv = 1.f / s;
#pragma unroll
  for (int i = 0; i < 8; ++i) x[i] *= inv;
}

__global__ void k_concat_out(const float* __restrict__ y, const float* __restrict__ ph,
                             float* __restrict__ out) {
  int row = blockIdx.x;
  int c = threadIdx.x;
  if (c < 132)       out[(size_t)row * 148 + c] = y[(size_t)row * 132 + c];
  else if (c < 148)  out[(size_t)row * 148 + c] = ph[(size_t)row * 16 + (c - 132)];
}

// ---------------------------------------------------------------------------
// Host orchestration
// ---------------------------------------------------------------------------
extern "C" void kernel_launch(void* const* d_in, const int* in_sizes, int n_in,
                              void* d_out, int out_size, void* d_ws, size_t ws_size,
                              hipStream_t stream) {
  (void)in_sizes; (void)n_in; (void)out_size; (void)ws_size;
  const float* motion = (const float*)d_in[0];
  const float* phase  = (const float*)d_in[1];
  const int*   mt     = (const int*)d_in[2];
  const float* enc_w1 = (const float*)d_in[3];
  const float* enc_b1 = (const float*)d_in[4];
  const float* enc_a1 = (const float*)d_in[5];
  const float* enc_w2 = (const float*)d_in[6];
  const float* enc_b2 = (const float*)d_in[7];
  const float* enc_a2 = (const float*)d_in[8];
  const float* rp_w1  = (const float*)d_in[9];
  const float* rp_b1  = (const float*)d_in[10];
  const float* rp_a   = (const float*)d_in[11];
  const float* rp_w2  = (const float*)d_in[12];
  const float* rp_b2  = (const float*)d_in[13];
  const float* att_qkv_w = (const float*)d_in[14];
  const float* att_qkv_b = (const float*)d_in[15];
  const float* att_o_w   = (const float*)d_in[16];
  const float* att_o_b   = (const float*)d_in[17];
  const float* att_ln_g  = (const float*)d_in[18];
  const float* att_ln_b  = (const float*)d_in[19];
  const float* pff_w1    = (const float*)d_in[20];
  const float* pff_b1    = (const float*)d_in[21];
  const float* pff_w2    = (const float*)d_in[22];
  const float* pff_b2    = (const float*)d_in[23];
  const float* pff_ln_g  = (const float*)d_in[24];
  const float* pff_ln_b  = (const float*)d_in[25];
  const float* final_ln_g = (const float*)d_in[26];
  const float* final_ln_b = (const float*)d_in[27];
  const float* pd_w1 = (const float*)d_in[28];
  const float* pd_b1 = (const float*)d_in[29];
  const float* pd_a  = (const float*)d_in[30];
  const float* pd_w2 = (const float*)d_in[31];
  const float* pd_b2 = (const float*)d_in[32];
  const float* g_w1  = (const float*)d_in[33];
  const float* g_b1  = (const float*)d_in[34];
  const float* g_a1  = (const float*)d_in[35];
  const float* g_w2  = (const float*)d_in[36];
  const float* g_b2  = (const float*)d_in[37];
  const float* g_a2  = (const float*)d_in[38];
  const float* g_w3  = (const float*)d_in[39];
  const float* g_b3  = (const float*)d_in[40];
  const float* ml_w0 = (const float*)d_in[41];
  const float* ml_b0 = (const float*)d_in[42];
  const float* ml_a0 = (const float*)d_in[43];
  const float* ml_w1 = (const float*)d_in[44];
  const float* ml_b1 = (const float*)d_in[45];
  const float* ml_a1 = (const float*)d_in[46];
  const float* ml_w2 = (const float*)d_in[47];
  const float* ml_b2 = (const float*)d_in[48];

  const int M = 8 * 512;  // 4096 tokens
  float* ws = (float*)d_ws;
  size_t off = 0;
  auto alloc = [&](size_t n) { float* p = ws + off; off += (n + 63) & ~(size_t)63; return p; };
  float* xin = alloc((size_t)M * XIN_LD);
  float* x   = alloc((size_t)M * 512);
  float* h   = alloc((size_t)M * 512);
  float* qkv = alloc((size_t)M * 2048);              // qkv / pff-mid / gating temps
  float* sc  = alloc((size_t)CH * 512 * 512);        // attention scores chunk / moe z1
  float* G   = alloc((size_t)CH * 512 * GS);         // Q.R^T chunk
  float* o   = alloc((size_t)M * 512);               // attn out / pd-mid / moe z0
  float* rr  = alloc((size_t)NPOS * 64);
  float* ph  = alloc((size_t)M * 16);
  float* gw  = alloc((size_t)M * 8);
  float* y   = alloc((size_t)M * 132);

  auto gemm = [&](const float* A, const float* W, const float* bias, const float* resid,
                  const float* alpha, float* C, int Mm, int Nn, int Kk, int lda, int Ka, int act) {
    dim3 grid((Nn + BN - 1) / BN, (Mm + BM - 1) / BM);
    bool full = (Mm % BM == 0) && (Nn % BN == 0) && (Kk % BK == 0) && (lda == Kk) && (Kk % 2 == 0);
    if (full)
      k_gemm_t<true><<<grid, 128, 0, stream>>>(A, W, bias, resid, alpha, C, Mm, Nn, Kk, lda, Ka, act);
    else
      k_gemm_t<false><<<grid, 128, 0, stream>>>(A, W, bias, resid, alpha, C, Mm, Nn, Kk, lda, Ka, act);
  };

  // 1. input assembly (mask + concat, zero-padded to XIN_LD)
  k_build_xin<<<M, XIN_LD, 0, stream>>>(motion, phase, mt, xin);

  // 2. encoder (A side padded: lda=Ka=160, logical K=145)
  gemm(xin, enc_w1, enc_b1, nullptr, enc_a1, h, M, 512, 145, XIN_LD, XIN_LD, 2);
  gemm(h,   enc_w2, enc_b2, nullptr, enc_a2, x, M, 512, 512, 512, 512, 2);

  // 3. relative position embeddings
  k_relpos<<<NPOS, 64, 0, stream>>>(rp_w1, rp_b1, rp_a, rp_w2, rp_b2, rr);

  // 4. transformer layers
  for (int i = 0; i < 4; ++i) {
    k_layernorm<<<M, 128, 0, stream>>>(x, att_ln_g + i * 512, att_ln_b + i * 512, h, 512);
    gemm(h, att_qkv_w + (size_t)i * 512 * 1536, att_qkv_b + i * 1536,
         nullptr, nullptr, qkv, M, 1536, 512, 512, 512, 0);
    for (int c = 0; c < 64 / CH; ++c) {
      int bh0 = c * CH;
      k_bd_gemm<<<dim3(16, 8, CH), 128, 0, stream>>>(qkv, rr, G, bh0);
      k_attn_scores<<<dim3(8, 8, CH), 128, 0, stream>>>(qkv, G, sc, bh0);
      k_softmax_rows<<<CH * 512, 128, 0, stream>>>(sc, 512, 0.125f);
      k_attn_av<<<dim3(1, 8, CH), 128, 0, stream>>>(sc, qkv, o, bh0);
    }
    gemm(o, att_o_w + (size_t)i * 512 * 512, att_o_b + i * 512, x, nullptr, x,
         M, 512, 512, 512, 512, 0);
    k_layernorm<<<M, 128, 0, stream>>>(x, pff_ln_g + i * 512, pff_ln_b + i * 512, h, 512);
    gemm(h,   pff_w1 + (size_t)i * 512 * 2048, pff_b1 + i * 2048, nullptr, nullptr, qkv,
         M, 2048, 512, 512, 512, 1);
    gemm(qkv, pff_w2 + (size_t)i * 2048 * 512, pff_b2 + i * 512,  x,       nullptr, x,
         M, 512, 2048, 2048, 2048, 0);
  }

  // 5. final LN (h = LN(x) feeds phase decoder and MoE)
  k_layernorm<<<M, 128, 0, stream>>>(x, final_ln_g, final_ln_b, h, 512);

  // 6. phase decoder (residual on phase)
  gemm(h, pd_w1, pd_b1, nullptr, pd_a, o, M, 512, 512, 512, 512, 2);
  gemm(o, pd_w2, pd_b2, phase, nullptr, ph, M, 16, 512, 512, 512, 0);

  // 7. gating network + softmax
  float* g1 = qkv;
  float* g2 = qkv + (size_t)M * 128;
  gemm(ph, g_w1, g_b1, nullptr, g_a1, g1, M, 128, 16, 16, 16, 2);
  gemm(g1, g_w2, g_b2, nullptr, g_a2, g2, M, 128, 128, 128, 128, 2);
  gemm(g2, g_w3, g_b3, nullptr, nullptr, gw, M, 8, 128, 128, 128, 0);
  k_softmax8<<<(M + 127) / 128, 128, 0, stream>>>(gw, M);

  // 8. MoE decoder (gated blend of 8 experts, fused)
  k_moe_t<true><<<dim3(8, 64), 128, 0, stream>>>(h,  ml_w0, ml_b0, gw, ml_a0, 2, o,  M, 512, 512, 8);
  k_moe_t<true><<<dim3(8, 64), 128, 0, stream>>>(o,  ml_w1, ml_b1, gw, ml_a1, 2, sc, M, 512, 512, 8);
  k_moe_t<false><<<dim3(3, 64), 128, 0, stream>>>(sc, ml_w2, ml_b2, gw, nullptr, 0, y, M, 132, 512, 8);

  // 9. concat [motion+contact | phase]
  k_concat_out<<<M, XIN_LD, 0, stream>>>(y, ph, (float*)d_out);
}
